// GPTOSSMoeBlock_27547920237065
// MI455X (gfx1250) — matrix operand order repaired and checked by reading, not compiled, not run on hardware
//
#include <hip/hip_runtime.h>

// ---------------- problem constants ----------------
#define B_TOK   512
#define H_DIM   1024
#define I_DIM   1024
#define TWO_I   2048
#define E_EXP   32
#define TOPK    4
#define EPS_C   1e-5f
#define ALPHA_C 1.702f
#define LIMIT_C 7.0f

// ---------------- CDNA5 WMMA types ----------------
typedef __attribute__((ext_vector_type(16))) __bf16        v16bf;
typedef __attribute__((ext_vector_type(8)))  float         v8f;
typedef __attribute__((ext_vector_type(4)))  unsigned int  u32x4;
typedef __attribute__((ext_vector_type(4)))  float         f32x4;

union Frag16 {
    v16bf          v;
    u32x4          q[2];
    unsigned short u[16];
};

// fp32 -> bf16 bits, round-to-nearest-even (scalar path, epilogues only)
__device__ __forceinline__ unsigned short f2bf(float f) {
    unsigned u = __float_as_uint(f);
    unsigned r = u + 0x7FFFu + ((u >> 16) & 1u);
    return (unsigned short)(r >> 16);
}

// two fp32 -> packed bf16x2 in ONE v_perm_b32 (+2 cheap adds for rounding)
__device__ __forceinline__ unsigned pkbf2(float lo, float hi) {
    unsigned a = __float_as_uint(lo) + 0x8000u;   // round-half-up
    unsigned b = __float_as_uint(hi) + 0x8000u;
    // bytes: [a.b2, a.b3, b.b2, b.b3] -> {bf16(hi), bf16(lo)}
    return __builtin_amdgcn_perm(b, a, 0x07060302u);
}

// convert 16 consecutive fp32 (two adjacent f32x4 pairs) into a bf16 B-frag
__device__ __forceinline__ void cvt_bfrag(Frag16& fr, const f32x4* bp) {
    const f32x4 b0 = bp[0], b1 = bp[1], b2 = bp[2], b3 = bp[3];
    fr.q[0].x = pkbf2(b0.x, b0.y);
    fr.q[0].y = pkbf2(b0.z, b0.w);
    fr.q[0].z = pkbf2(b1.x, b1.y);
    fr.q[0].w = pkbf2(b1.z, b1.w);
    fr.q[1].x = pkbf2(b2.x, b2.y);
    fr.q[1].y = pkbf2(b2.z, b2.w);
    fr.q[1].z = pkbf2(b3.x, b3.y);
    fr.q[1].w = pkbf2(b3.z, b3.w);
}

// ---------------- kernel 0: zero per-expert counters ----------------
__global__ void zero_counts_kernel(int* __restrict__ counts) {
    if (threadIdx.x < E_EXP) counts[threadIdx.x] = 0;
}

// ---------------- kernel 1: rmsnorm + gate + routing + residual ----------------
__global__ __launch_bounds__(256)
void routing_kernel(const float* __restrict__ x,
                    const float* __restrict__ norm_scale,
                    const float* __restrict__ gate_w,
                    const float* __restrict__ gate_b,
                    unsigned short* __restrict__ tbf,   // [B,H] bf16 normed act
                    int*   __restrict__ counts,         // [E]
                    int*   __restrict__ tlist,          // [E,B]
                    float* __restrict__ twt,            // [E,B]
                    float* __restrict__ out)            // [B,H] init = x
{
    const int b   = blockIdx.x;
    const int tid = threadIdx.x;

    __shared__ float sred[256];
    __shared__ float tvals[H_DIM];
    __shared__ float gpart[256];
    __shared__ float glog[E_EXP];

    const f32x4 xv = ((const f32x4*)(x + (size_t)b * H_DIM))[tid];
    float ss = xv.x * xv.x + xv.y * xv.y + xv.z * xv.z + xv.w * xv.w;
    sred[tid] = ss;
    __syncthreads();
    for (int s = 128; s >= 1; s >>= 1) {
        if (tid < s) sred[tid] += sred[tid + s];
        __syncthreads();
    }
    const float rms = rsqrtf(sred[0] * (1.0f / (float)H_DIM) + EPS_C);

    ((f32x4*)(out + (size_t)b * H_DIM))[tid] = xv;   // residual init
    {
        const int h0 = tid * 4;
        float t0 = xv.x * rms * norm_scale[h0 + 0];
        float t1 = xv.y * rms * norm_scale[h0 + 1];
        float t2 = xv.z * rms * norm_scale[h0 + 2];
        float t3 = xv.w * rms * norm_scale[h0 + 3];
        tvals[h0 + 0] = t0; tvals[h0 + 1] = t1;
        tvals[h0 + 2] = t2; tvals[h0 + 3] = t3;
        unsigned* tb = (unsigned*)(tbf + (size_t)b * H_DIM + h0);
        tb[0] = pkbf2(t0, t1);
        tb[1] = pkbf2(t2, t3);
    }
    __syncthreads();

    // gate logits: thread = (part 0..7) x (expert 0..31)
    {
        const int e    = tid & 31;
        const int part = tid >> 5;
        const float* gw = gate_w + (size_t)e * H_DIM + part * 128;
        float acc = 0.f;
        #pragma unroll 8
        for (int h = 0; h < 128; ++h) acc += tvals[part * 128 + h] * gw[h];
        gpart[tid] = acc;
    }
    __syncthreads();
    if (tid < E_EXP) {
        float s = gate_b[tid];
        #pragma unroll
        for (int p = 0; p < 8; ++p) s += gpart[p * 32 + tid];
        glog[tid] = s;
    }
    __syncthreads();

    if (tid == 0) {
        float bv[TOPK]; int bi[TOPK];
        #pragma unroll
        for (int k = 0; k < TOPK; ++k) {
            float best = -__builtin_inff(); int bidx = 0;
            for (int e = 0; e < E_EXP; ++e) {
                float v = glog[e];
                if (v > best) { best = v; bidx = e; }
            }
            bv[k] = best; bi[k] = bidx;
            glog[bidx] = -__builtin_inff();
        }
        const float m = bv[0];
        float den = 0.f, ex[TOPK];
        #pragma unroll
        for (int k = 0; k < TOPK; ++k) { ex[k] = __expf(bv[k] - m); den += ex[k]; }
        const float inv = 1.0f / den;
        #pragma unroll
        for (int k = 0; k < TOPK; ++k) {
            const int e   = bi[k];
            const int pos = atomicAdd(&counts[e], 1);
            tlist[e * B_TOK + pos] = b;
            twt[e * B_TOK + pos]   = ex[k] * inv;
        }
    }
}

// ---------------- kernel 2: grouped expert MLP via WMMA bf16 ----------------
// grid: (E, maxTiles=32), block: 256 threads = 8 waves.
// Each block: expert e, 16-token M-tile.
__global__ __launch_bounds__(256)
void expert_kernel(const unsigned short* __restrict__ tbf,  // [B,H] bf16
                   const int*   __restrict__ counts,
                   const int*   __restrict__ tlist,
                   const float* __restrict__ twt,
                   const float* __restrict__ mlp1_w,        // [E,2I,H]
                   const float* __restrict__ mlp1_b,        // [E,2I]
                   const float* __restrict__ mlp2_w,        // [E,H,I]
                   const float* __restrict__ mlp2_b,        // [E,H]
                   float* __restrict__ out)                 // [B,H] (+=, atomic)
{
    const int e    = blockIdx.x;
    const int tile = blockIdx.y;
    const int cnt  = counts[e];
    if (tile * 16 >= cnt) return;

    __shared__ int   tok[16];
    __shared__ float wt[16];
    __shared__ unsigned short sA[16 * I_DIM];   // swiglu output, bf16, 32 KB

    const int tid = threadIdx.x;
    if (tid < 16) {
        const int g = tile * 16 + tid;
        if (g < cnt) { tok[tid] = tlist[e * B_TOK + g]; wt[tid] = twt[e * B_TOK + g]; }
        else         { tok[tid] = 0;                    wt[tid] = 0.0f; }
    }
    __syncthreads();

    const int lane = tid & 31;
    const int wv   = tid >> 5;                 // wave id 0..7
    const int mrow = lane & 15;                // M row / B(N) column index
    const int hi   = lane >> 4;                // lane-half select
    const int kbA  = hi ? 16 : 0;              // A-frag K base (contiguous k = 16*hi + s)
    const int kbB  = hi ? 16 : 0;              // B-frag K base

    // ---------- phase 1: glu/lin rows as two B matrices sharing one A-frag ----
    // a[:, c] = swiglu(h[:, 2c], h[:, 2c+1]); lane owns a-column c = c0 + mrow.
    const unsigned short* arow = tbf + (size_t)tok[mrow] * H_DIM;
    for (int ct = wv; ct < I_DIM / 16; ct += 8) {
        const int c0 = ct * 16;
        const int cg = 2 * (c0 + mrow);                              // glu row of W1
        const float* wg = mlp1_w + ((size_t)e * TWO_I + cg) * H_DIM; // row 2c
        const float* wl = wg + H_DIM;                                // row 2c+1
        v8f accG = {}, accL = {};
        #pragma unroll 2
        for (int k0 = 0; k0 < H_DIM; k0 += 32) {
            Frag16 A, Bg, Bl;
            A.q[0] = *(const u32x4*)(arow + k0 + kbA);
            A.q[1] = *(const u32x4*)(arow + k0 + kbA + 8);
            __builtin_prefetch(wg + k0 + 64, 0, 1);
            __builtin_prefetch(wl + k0 + 64, 0, 1);
            cvt_bfrag(Bg, (const f32x4*)(wg + k0 + kbB));
            cvt_bfrag(Bl, (const f32x4*)(wl + k0 + kbB));
            accG = __builtin_amdgcn_wmma_f32_16x16x32_bf16(
                       false, A.v, false, Bg.v, (short)0, accG, false, false);
            accL = __builtin_amdgcn_wmma_f32_16x16x32_bf16(
                       false, A.v, false, Bl.v, (short)0, accL, false, false);
        }
        const float bg = mlp1_b[e * TWO_I + cg];
        const float bl = mlp1_b[e * TWO_I + cg + 1];
        #pragma unroll
        for (int r = 0; r < 8; ++r) {
            float g   = fminf(accG[r] + bg, LIMIT_C);
            float lin = fminf(fmaxf(accL[r] + bl, -LIMIT_C), LIMIT_C);
            float sig = __builtin_amdgcn_rcpf(1.0f + __expf(-ALPHA_C * g));
            float av  = g * sig * (lin + 1.0f);
            sA[(r + (hi << 3)) * I_DIM + c0 + mrow] = f2bf(av);
        }
    }
    __syncthreads();

    // ---------- phase 2: o = a @ W2^T + b2 ; out += wt * o (atomic) ----------
    const unsigned short* arow2 = sA + mrow * I_DIM;
    for (int nt = wv; nt < H_DIM / 16; nt += 8) {
        const int h0 = nt * 16;
        const float* wrow = mlp2_w + ((size_t)e * H_DIM + h0 + mrow) * I_DIM;
        v8f acc = {};
        #pragma unroll 2
        for (int k0 = 0; k0 < I_DIM; k0 += 32) {
            Frag16 A, Bf;
            A.q[0] = *(const u32x4*)(arow2 + k0 + kbA);        // ds_load_b128
            A.q[1] = *(const u32x4*)(arow2 + k0 + kbA + 8);
            __builtin_prefetch(wrow + k0 + 64, 0, 1);
            cvt_bfrag(Bf, (const f32x4*)(wrow + k0 + kbB));
            acc = __builtin_amdgcn_wmma_f32_16x16x32_bf16(
                      false, A.v, false, Bf.v, (short)0, acc, false, false);
        }
        const float bias = mlp2_b[e * H_DIM + h0 + mrow];
        #pragma unroll
        for (int r = 0; r < 8; ++r) {
            const int m = r + (hi << 3);
            const float val = (acc[r] + bias) * wt[m];
            atomicAdd(out + (size_t)tok[m] * H_DIM + (h0 + mrow), val);
        }
    }
}

// ---------------- launcher ----------------
extern "C" void kernel_launch(void* const* d_in, const int* in_sizes, int n_in,
                              void* d_out, int out_size, void* d_ws, size_t ws_size,
                              hipStream_t stream) {
    const float* x    = (const float*)d_in[0];
    const float* ns   = (const float*)d_in[1];
    const float* gw   = (const float*)d_in[2];
    const float* gb   = (const float*)d_in[3];
    const float* w1   = (const float*)d_in[4];
    const float* b1   = (const float*)d_in[5];
    const float* w2   = (const float*)d_in[6];
    const float* b2   = (const float*)d_in[7];
    float* out        = (float*)d_out;

    char* ws = (char*)d_ws;
    unsigned short* tbf = (unsigned short*)ws;                       // 1 MB
    int*   counts = (int*)  (ws + (size_t)B_TOK * H_DIM * 2);        // 128 B
    int*   tlist  = (int*)  (ws + (size_t)B_TOK * H_DIM * 2 + 256);  // 64 KB
    float* twt    = (float*)(ws + (size_t)B_TOK * H_DIM * 2 + 256 + (size_t)E_EXP * B_TOK * 4);

    zero_counts_kernel<<<dim3(1), dim3(32), 0, stream>>>(counts);
    routing_kernel<<<dim3(B_TOK), dim3(256), 0, stream>>>(
        x, ns, gw, gb, tbf, counts, tlist, twt, out);
    expert_kernel<<<dim3(E_EXP, B_TOK / 16), dim3(256), 0, stream>>>(
        tbf, counts, tlist, twt, w1, b1, w2, b2, out);
}
